// FusionCrossAttention_71210557768287
// MI455X (gfx1250) — matrix operand order repaired and checked by reading, not compiled, not run on hardware
//
#include <hip/hip_runtime.h>

typedef unsigned short ushort_t;
typedef unsigned int   uint_t;
typedef __attribute__((ext_vector_type(16))) __bf16  v16bf;
typedef __attribute__((ext_vector_type(8)))  __bf16  v8bf;
typedef __attribute__((ext_vector_type(8)))  float   v8f;
typedef __attribute__((ext_vector_type(4)))  float   f32x4;
typedef __attribute__((ext_vector_type(4)))  uint_t  u32x4;

#define DM 1024
#define NH 16
#define HD 64

// native f32->bf16 (RNE) via hardware cvt
__device__ __forceinline__ uint_t pack2(float a, float b) {
  union { __bf16 b[2]; uint_t u; } c;
  c.b[0] = (__bf16)a; c.b[1] = (__bf16)b;
  return c.u;
}
__device__ __forceinline__ v16bf concat8(v8bf lo, v8bf hi) {
  return __builtin_shufflevector(lo, hi, 0,1,2,3,4,5,6,7,8,9,10,11,12,13,14,15);
}
__device__ __forceinline__ v8f vsplat(float x) {
  v8f v;
#pragma unroll
  for (int i = 0; i < 8; i++) v[i] = x;
  return v;
}
__device__ __forceinline__ v8f wmma_bf16(v16bf a, v16bf b, v8f c) {
  return __builtin_amdgcn_wmma_f32_16x16x32_bf16(false, a, false, b, (short)0, c, false, false);
}
__device__ __forceinline__ void cvt_store16(ushort_t* dst, const f32x4 x[4]) {
  u32x4 p0, p1;
  p0[0] = pack2(x[0][0], x[0][1]); p0[1] = pack2(x[0][2], x[0][3]);
  p0[2] = pack2(x[1][0], x[1][1]); p0[3] = pack2(x[1][2], x[1][3]);
  p1[0] = pack2(x[2][0], x[2][1]); p1[1] = pack2(x[2][2], x[2][3]);
  p1[2] = pack2(x[3][0], x[3][1]); p1[3] = pack2(x[3][2], x[3][3]);
  ((u32x4*)dst)[0] = p0; ((u32x4*)dst)[1] = p1;
}

// hardware exp2 (v_exp_f32 is natively exp2)
__device__ __forceinline__ float fexp2(float x) {
#if __has_builtin(__builtin_amdgcn_exp2f)
  return __builtin_amdgcn_exp2f(x);
#else
  return exp2f(x);
#endif
}

// lane XOR within each 16-lane row: VALU v_permlane16_b32 (off the LDS pipe)
__device__ __forceinline__ float xor16(float x, uint_t s0, uint_t s1, int msk) {
#if __has_builtin(__builtin_amdgcn_permlane16)
  uint_t u = __builtin_bit_cast(uint_t, x);
  u = __builtin_amdgcn_permlane16(u, u, s0, s1, false, false);
  return __builtin_bit_cast(float, u);
#else
  return __shfl_xor(x, msk, 32);
#endif
}
// butterfly reductions over the 16 lanes of each half-wave
__device__ __forceinline__ void redmax16(v8f& v) {
#pragma unroll
  for (int i = 0; i < 8; i++) v[i] = fmaxf(v[i], xor16(v[i], 0x67452301u, 0xEFCDAB89u, 1));
#pragma unroll
  for (int i = 0; i < 8; i++) v[i] = fmaxf(v[i], xor16(v[i], 0x54761032u, 0xDCFE98BAu, 2));
#pragma unroll
  for (int i = 0; i < 8; i++) v[i] = fmaxf(v[i], xor16(v[i], 0x32107654u, 0xBA98FEDCu, 4));
#pragma unroll
  for (int i = 0; i < 8; i++) v[i] = fmaxf(v[i], xor16(v[i], 0xFEDCBA98u, 0x76543210u, 8));
}
__device__ __forceinline__ void redsum16(v8f& v) {
#pragma unroll
  for (int i = 0; i < 8; i++) v[i] += xor16(v[i], 0x67452301u, 0xEFCDAB89u, 1);
#pragma unroll
  for (int i = 0; i < 8; i++) v[i] += xor16(v[i], 0x54761032u, 0xDCFE98BAu, 2);
#pragma unroll
  for (int i = 0; i < 8; i++) v[i] += xor16(v[i], 0x32107654u, 0xBA98FEDCu, 4);
#pragma unroll
  for (int i = 0; i < 8; i++) v[i] += xor16(v[i], 0xFEDCBA98u, 0x76543210u, 8);
}

// ---------------------------------------------------------------------------
// y = oscale * (x @ W^T).  x: (M,Kd) fp32 or bf16, W: (N,Kd) fp32.
// Tile 128x128 / WG, 8 waves, double-buffered LDS, K-step 32.
// OUT_MODE: 0 = bf16 head-major (B,H,Tb,HD)
//           1 = bf16 head-major TRANSPOSED (B,H,HD,Tb)   (for V)
//           2 = fp32 (M,N)
// ---------------------------------------------------------------------------
template<bool A_BF16, int OUT_MODE>
__global__ __launch_bounds__(256) void gemm_xWT(
    const void* __restrict__ Ap, const float* __restrict__ W,
    void* __restrict__ Cp, int N, int Kd, int Tb, float oscale)
{
  __shared__ __align__(16) ushort_t Als[2][128 * 32];
  __shared__ __align__(16) ushort_t Bls[2][128 * 32];

  const int t    = threadIdx.x;
  const int lane = t & 31;
  const int wv   = t >> 5;
  const int lr   = lane & 15;
  const int hv   = lane >> 4;          // wave half
  const int wm   = (wv >> 2) * 64;
  const int wn   = (wv & 3) * 32;
  const int bm   = blockIdx.x * 128;
  const int bn   = blockIdx.y * 128;
  const int sr   = t >> 1;             // staging row 0..127
  const int sc   = (t & 1) * 16;       // staging col 0/16

  f32x4 ax[4]; u32x4 au[2]; f32x4 bx[4];

  auto fetch = [&](int k0) {
    if (A_BF16) {
      const u32x4* g = (const u32x4*)((const ushort_t*)Ap + (size_t)(bm + sr) * Kd + k0 + sc);
      au[0] = g[0]; au[1] = g[1];
    } else {
      const f32x4* g = (const f32x4*)((const float*)Ap + (size_t)(bm + sr) * Kd + k0 + sc);
      ax[0] = g[0]; ax[1] = g[1]; ax[2] = g[2]; ax[3] = g[3];
    }
    const f32x4* gw = (const f32x4*)(W + (size_t)(bn + sr) * Kd + k0 + sc);
    bx[0] = gw[0]; bx[1] = gw[1]; bx[2] = gw[2]; bx[3] = gw[3];
  };
  auto stage = [&](int buf) {
    if (A_BF16) {
      u32x4* d = (u32x4*)&Als[buf][sr * 32 + sc];
      d[0] = au[0]; d[1] = au[1];
    } else {
      cvt_store16(&Als[buf][sr * 32 + sc], ax);
    }
    cvt_store16(&Bls[buf][sr * 32 + sc], bx);
  };

  v8f acc[4][2];
#pragma unroll
  for (int i = 0; i < 4; i++)
#pragma unroll
    for (int j = 0; j < 2; j++) acc[i][j] = vsplat(0.f);

  fetch(0);
  stage(0);
  __syncthreads();
  int cur = 0;

  for (int k0 = 0; k0 < Kd; k0 += 32) {
    const bool more = (k0 + 32) < Kd;
    if (more) fetch(k0 + 32);          // issue next tile's global loads early

    // A fragments: lane holds row lr, 16 contiguous K at kb=16*half
    v16bf af[4];
#pragma unroll
    for (int mi = 0; mi < 4; mi++) {
      const int r  = wm + mi * 16 + lr;
      const int kb = hv * 16;
      v8bf lo = *(const v8bf*)&Als[cur][r * 32 + kb];
      v8bf hi = *(const v8bf*)&Als[cur][r * 32 + kb + 8];
      af[mi] = concat8(lo, hi);
    }
    // B fragments: lane holds col lr, 16 contiguous K at 16*half
    v16bf bfr[2];
#pragma unroll
    for (int ni = 0; ni < 2; ni++) {
      const int r = wn + ni * 16 + lr;
      bfr[ni] = *(const v16bf*)&Bls[cur][r * 32 + hv * 16];
    }
#pragma unroll
    for (int mi = 0; mi < 4; mi++)
#pragma unroll
      for (int ni = 0; ni < 2; ni++)
        acc[mi][ni] = wmma_bf16(af[mi], bfr[ni], acc[mi][ni]);

    if (more) stage(cur ^ 1);
    cur ^= 1;
    __syncthreads();
  }

  // epilogue: C layout lane=col, VGPR r -> row r + 8*half
#pragma unroll
  for (int mi = 0; mi < 4; mi++)
#pragma unroll
    for (int ni = 0; ni < 2; ni++)
#pragma unroll
      for (int r = 0; r < 8; r++) {
        const int row = bm + wm + mi * 16 + r + hv * 8;
        const int col = bn + wn + ni * 16 + lr;
        const float v = acc[mi][ni][r] * oscale;
        if (OUT_MODE == 2) {
          ((float*)Cp)[(size_t)row * N + col] = v;
        } else {
          const int b  = row / Tb;
          const int tt = row - b * Tb;
          size_t idx;
          if (OUT_MODE == 0) {
            const int h = col >> 6, dd = col & 63;
            idx = (((size_t)b * NH + h) * Tb + tt) * HD + dd;
          } else {
            idx = ((size_t)b * DM + col) * Tb + tt;   // V^T: (B,H,HD,Tb)
          }
          ((__bf16*)Cp)[idx] = (__bf16)v;
        }
      }
}

// ---------------------------------------------------------------------------
// Flash attention: 4 waves / block, 64 q rows; kv chunks of 64, double-buffered.
// Q is PRE-SCALED by 0.125*log2(e) -> softmax runs in exp2 domain.
// Q,K bf16 (B,H,T,HD); V bf16 TRANSPOSED (B,H,HD,Tkv). Out bf16 (B,Tq,H,HD).
// ---------------------------------------------------------------------------
__global__ __launch_bounds__(128) void attn_flash(
    const ushort_t* __restrict__ Q, const ushort_t* __restrict__ K,
    const ushort_t* __restrict__ V, ushort_t* __restrict__ O,
    int Tq, int Tkv)
{
  __shared__ __align__(16) ushort_t Kls[2][64 * 64];   // [kv][d]
  __shared__ __align__(16) ushort_t Vt [2][64 * 72];   // [d][kv] pad 72
  __shared__ __align__(16) ushort_t Pls[4][16 * 64];   // per-wave P strip

  const int t    = threadIdx.x;
  const int lane = t & 31;
  const int wv   = t >> 5;
  const int lr   = lane & 15;
  const int hv   = lane >> 4;
  const int bh   = blockIdx.x;
  const int qt   = blockIdx.y;

  // Q A-fragments straight from global
  const size_t qbase = ((size_t)bh * Tq + qt * 64 + wv * 16 + lr) * HD;
  v16bf qa[2];
#pragma unroll
  for (int s = 0; s < 2; s++) {
    v8bf lo = *(const v8bf*)(Q + qbase + s * 32 + hv * 16);
    v8bf hi = *(const v8bf*)(Q + qbase + s * 32 + hv * 16 + 8);
    qa[s] = concat8(lo, hi);
  }

  // staging: thread -> K row sr2 (kv), cols sc2..+31 (d); V row sr2 (d), cols (kv)
  const int sr2 = t >> 1;
  const int sc2 = (t & 1) * 32;
  const size_t kgbase = ((size_t)bh * Tkv + sr2) * HD + sc2;
  const size_t vgbase = ((size_t)bh * HD + sr2) * Tkv + sc2;

  u32x4 krg[4], vrg[4];
  auto fetch = [&](int kv0) {
    const u32x4* gk = (const u32x4*)(K + kgbase + (size_t)kv0 * HD);
    krg[0] = gk[0]; krg[1] = gk[1]; krg[2] = gk[2]; krg[3] = gk[3];
    const u32x4* gv = (const u32x4*)(V + vgbase + kv0);
    vrg[0] = gv[0]; vrg[1] = gv[1]; vrg[2] = gv[2]; vrg[3] = gv[3];
  };
  auto stage = [&](int buf) {
    u32x4* dk = (u32x4*)&Kls[buf][sr2 * 64 + sc2];
    dk[0] = krg[0]; dk[1] = krg[1]; dk[2] = krg[2]; dk[3] = krg[3];
    u32x4* dv = (u32x4*)&Vt[buf][sr2 * 72 + sc2];
    dv[0] = vrg[0]; dv[1] = vrg[1]; dv[2] = vrg[2]; dv[3] = vrg[3];
  };

  v8f o[4];
#pragma unroll
  for (int jd = 0; jd < 4; jd++) o[jd] = vsplat(0.f);
  v8f mrow = vsplat(-3.0e38f);
  v8f lrow = vsplat(0.f);

  fetch(0);
  stage(0);
  __syncthreads();
  int cur = 0;
  const int nc = Tkv / 64;

  for (int c = 0; c < nc; ++c) {
    if (c + 1 < nc) fetch((c + 1) * 64);

    // ---- S = Q K^T (already in log2 domain) : 4 tiles, 2 wmma each ----
    v8f s4[4];
#pragma unroll
    for (int j = 0; j < 4; j++) {
      const v16bf b0 = *(const v16bf*)&Kls[cur][(j * 16 + lr) * 64 + hv * 16];
      const v16bf b1 = *(const v16bf*)&Kls[cur][(j * 16 + lr) * 64 + 32 + hv * 16];
      v8f a = vsplat(0.f);
      a = wmma_bf16(qa[0], b0, a);
      a = wmma_bf16(qa[1], b1, a);
      s4[j] = a;
    }

    // ---- online softmax (exp2 domain; row stats across 16-lane halves) ----
    v8f mloc = s4[0];
#pragma unroll
    for (int j = 1; j < 4; j++)
#pragma unroll
      for (int i = 0; i < 8; i++) mloc[i] = fmaxf(mloc[i], s4[j][i]);
    redmax16(mloc);

    v8f mnew, alpha;
#pragma unroll
    for (int i = 0; i < 8; i++) {
      mnew[i]  = fmaxf(mrow[i], mloc[i]);
      alpha[i] = fexp2(mrow[i] - mnew[i]);
    }
    v8f psum = vsplat(0.f);
#pragma unroll
    for (int j = 0; j < 4; j++)
#pragma unroll
      for (int i = 0; i < 8; i++) {
        float p = fexp2(s4[j][i] - mnew[i]);
        s4[j][i] = p;
        psum[i] += p;
      }
    redsum16(psum);
#pragma unroll
    for (int i = 0; i < 8; i++) {
      lrow[i] = lrow[i] * alpha[i] + psum[i];
      mrow[i] = mnew[i];
    }
#pragma unroll
    for (int jd = 0; jd < 4; jd++)
#pragma unroll
      for (int i = 0; i < 8; i++) o[jd][i] *= alpha[i];

    // ---- P (C-layout) -> wave-private LDS strip ----
#pragma unroll
    for (int j = 0; j < 4; j++)
#pragma unroll
      for (int r = 0; r < 8; r++)
        ((__bf16*)Pls[wv])[(r + hv * 8) * 64 + j * 16 + lr] = (__bf16)s4[j][r];

    // wave-private LDS: in-order per wave -> counter wait suffices (no barrier)
    asm volatile("s_wait_dscnt 0x0" ::: "memory");

    v16bf pa[2];
#pragma unroll
    for (int s = 0; s < 2; s++) {
      v8bf lo = *(const v8bf*)&Pls[wv][lr * 64 + s * 32 + hv * 16];
      v8bf hi = *(const v8bf*)&Pls[wv][lr * 64 + s * 32 + hv * 16 + 8];
      pa[s] = concat8(lo, hi);
    }

    // ---- O += P V : B-fragments from Vt (already transposed in global) ----
#pragma unroll
    for (int jd = 0; jd < 4; jd++) {
      const int vb = (jd * 16 + lr) * 72;
#pragma unroll
      for (int s = 0; s < 2; s++) {
        const int kb = s * 32 + hv * 16;
        v8bf lo = *(const v8bf*)&Vt[cur][vb + kb];
        v8bf hi = *(const v8bf*)&Vt[cur][vb + kb + 8];
        o[jd] = wmma_bf16(pa[s], concat8(lo, hi), o[jd]);
      }
    }

    if (c + 1 < nc) stage(cur ^ 1);
    cur ^= 1;
    __syncthreads();
  }

  // ---- epilogue: O *= 1/l, store bf16 as (B,Tq,H,HD) ----
  v8f inv;
#pragma unroll
  for (int i = 0; i < 8; i++) inv[i] = 1.0f / lrow[i];
  const int b = bh >> 4, h = bh & 15;
#pragma unroll
  for (int jd = 0; jd < 4; jd++)
#pragma unroll
    for (int r = 0; r < 8; r++) {
      const int qq = qt * 64 + wv * 16 + r + hv * 8;
      const int dd = jd * 16 + lr;
      const float v = o[jd][r] * inv[r];
      ((__bf16*)O)[(((size_t)b * Tq + qq) * NH + h) * HD + dd] = (__bf16)v;
    }
}

// ---------------------------------------------------------------------------
extern "C" void kernel_launch(void* const* d_in, const int* in_sizes, int n_in,
                              void* d_out, int out_size, void* d_ws, size_t ws_size,
                              hipStream_t stream) {
  const float* query = (const float*)d_in[0];
  const float* kvsrc = (const float*)d_in[1];
  const float* wq    = (const float*)d_in[2];
  const float* wk    = (const float*)d_in[3];
  const float* wv    = (const float*)d_in[4];
  const float* wo    = (const float*)d_in[5];
  const int B = 4, Tq = 1024, Tkv = 4096;

  // workspace: Q (8MB) | K (32MB) | V^T (32MB) | attn-out (8MB), all bf16
  ushort_t* qws = (ushort_t*)d_ws;
  ushort_t* kws = qws + (size_t)B * DM * Tq;
  ushort_t* vws = kws + (size_t)B * DM * Tkv;
  ushort_t* aws = vws + (size_t)B * DM * Tkv;

  // fold softmax scale AND log2(e) into Q so attention uses raw v_exp_f32 (exp2)
  const float qscale = 0.125f * 1.4426950408889634f;

  dim3 blk(256);
  gemm_xWT<false, 0><<<dim3((B * Tq)  / 128, DM / 128), blk, 0, stream>>>(query, wq, qws, DM, DM, Tq,  qscale);
  gemm_xWT<false, 0><<<dim3((B * Tkv) / 128, DM / 128), blk, 0, stream>>>(kvsrc, wk, kws, DM, DM, Tkv, 1.0f);
  gemm_xWT<false, 1><<<dim3((B * Tkv) / 128, DM / 128), blk, 0, stream>>>(kvsrc, wv, vws, DM, DM, Tkv, 1.0f);
  attn_flash<<<dim3(B * NH, Tq / 64), dim3(128), 0, stream>>>(qws, kws, vws, aws, Tq, Tkv);
  gemm_xWT<true, 2><<<dim3((B * Tq) / 128, DM / 128), blk, 0, stream>>>(aws, wo, d_out, DM, DM, 1, 1.0f);
}
